// DeepCorNN_75350906241094
// MI455X (gfx1250) — compile-verified, matrix operand order checked by reading, not run on hardware
//
#include <hip/hip_runtime.h>
#include <hip/hip_bf16.h>
#include <math.h>

#define NFEAT 25
#define DIMV  512
#define HIDV  256
#define BATCHV 4096
#define NBLK  11
#define EPSV  1e-5f

typedef __attribute__((ext_vector_type(16))) __bf16 v16bf;
typedef __attribute__((ext_vector_type(8)))  float  v8f;
typedef __attribute__((ext_vector_type(4)))  int    v4i;

#if __has_builtin(__builtin_amdgcn_global_load_async_to_lds_b128) && \
    __has_builtin(__builtin_amdgcn_s_wait_asynccnt)
#define HAS_ASYNC_LDS 1
#else
#define HAS_ASYNC_LDS 0
#endif

// ---------- small helpers ----------
__device__ inline unsigned short f2bf(float f) {
  union { float f; unsigned int u; } v; v.f = f;
  unsigned int r = v.u + 0x7FFFu + ((v.u >> 16) & 1u);
  return (unsigned short)(r >> 16);
}
__device__ inline __bf16 us2bf(unsigned short u) {
  union { unsigned short u; __bf16 b; } c; c.u = u; return c.b;
}
__device__ inline v8f zero8() {
  v8f v;
#pragma unroll
  for (int i = 0; i < 8; ++i) v[i] = 0.f;
  return v;
}
__device__ inline v8f wmma_bf16(v16bf a, v16bf b, v8f c) {
  return __builtin_amdgcn_wmma_f32_16x16x32_bf16(false, a, false, b, (short)0, c,
                                                 false, false);
}
__device__ inline float ssign(float x) { return x / (1.0f + fabsf(x)); }

// async global->LDS 16B per lane (CDNA5 GLOBAL_LOAD_ASYNC_TO_LDS_B128)
__device__ inline void async_g2l_b128(const void* g, void* l) {
#if HAS_ASYNC_LDS
  __builtin_amdgcn_global_load_async_to_lds_b128(
      (__attribute__((address_space(1))) v4i*)g,
      (__attribute__((address_space(3))) v4i*)l, 0, 0);
#else
  (void)g; (void)l;
#endif
}
__device__ inline void wait_async0() {
#if HAS_ASYNC_LDS
  __builtin_amdgcn_s_wait_asynccnt(0);
#endif
}

// A fragment (16-bit, 16x32): lane holds row M=lane%16; K = e + (e&8) + 8*(lane>=16)
__device__ inline v16bf frag_rowmajor(const unsigned short* lds, int row0, int stride) {
  const int lane = threadIdx.x & 31;
  const int hi = (lane >> 4) & 1;
  const int r = row0 + (lane & 15);
  v16bf f;
#pragma unroll
  for (int e = 0; e < 16; ++e) {
    int k = e + (e & 8) + (hi << 3);
    f[e] = us2bf(lds[r * stride + k]);
  }
  return f;
}
// B fragment (16-bit, 32x16, B stored row-major K x N): lane holds col N=lane%16
__device__ inline v16bf frag_colmajor(const unsigned short* lds, int col0, int stride) {
  const int lane = threadIdx.x & 31;
  const int hi = (lane >> 4) & 1;
  const int n = col0 + (lane & 15);
  v16bf f;
#pragma unroll
  for (int e = 0; e < 16; ++e) {
    int k = e + (e & 8) + (hi << 3);
    f[e] = us2bf(lds[k * stride + n]);
  }
  return f;
}

// ---------- weight pack fp32 -> bf16 ----------
__global__ void pack_k(const float* __restrict__ W0, const float* __restrict__ W1,
                       unsigned short* __restrict__ W0b, unsigned short* __restrict__ W1b,
                       size_t n) {
  for (size_t i = (size_t)blockIdx.x * blockDim.x + threadIdx.x; i < n;
       i += (size_t)gridDim.x * blockDim.x) {
    W0b[i] = f2bf(W0[i]);
    W1b[i] = f2bf(W1[i]);
  }
}

__global__ void zero_k(float* p, int n) {
  if ((int)threadIdx.x < n) p[threadIdx.x] = 0.f;
}

// ---------- embedding + LayerNorm ----------
__global__ void embed_k(const int* __restrict__ x, const float* __restrict__ emb,
                        const float* __restrict__ loc, const float* __restrict__ g,
                        const float* __restrict__ b, float* __restrict__ X) {
  const int row = blockIdx.x;           // n*25 + c
  const int c = row % NFEAT;
  const int tok = x[row];
  __shared__ float buf[DIMV];
  __shared__ float rs[256];
  __shared__ float rq[256];
  const int t = threadIdx.x;
  float v0 = emb[(size_t)tok * DIMV + t] + loc[(size_t)c * DIMV + t];
  float v1 = emb[(size_t)tok * DIMV + 256 + t] + loc[(size_t)c * DIMV + 256 + t];
  buf[t] = v0; buf[256 + t] = v1;
  rs[t] = v0 + v1; rq[t] = v0 * v0 + v1 * v1;
  __syncthreads();
  for (int s = 128; s > 0; s >>= 1) {
    if (t < s) { rs[t] += rs[t + s]; rq[t] += rq[t + s]; }
    __syncthreads();
  }
  float m = rs[0] * (1.0f / DIMV);
  float var = rq[0] * (1.0f / DIMV) - m * m;
  float inv = rsqrtf(var + EPSV);
  X[(size_t)row * DIMV + t]        = (v0 - m) * inv * g[t] + b[t];
  X[(size_t)row * DIMV + 256 + t]  = (v1 - m) * inv * g[256 + t] + b[256 + t];
}

// ---------- WMMA GEMM: C(MxN) = A(MxK) * B(KxN) + bias ----------
// BM=128 BN=64 BK=32, 256 threads = 8 waves in a 4x2 grid, 32x32 per wave.
// B tile (and A tile when A is bf16) staged via GLOBAL_LOAD_ASYNC_TO_LDS_B128.
__global__ void gemm_k(const float* __restrict__ A32, const unsigned short* __restrict__ A16,
                       const unsigned short* __restrict__ B, const float* __restrict__ bias,
                       float* __restrict__ C, int M, int K, int Nout) {
  __shared__ __attribute__((aligned(16))) unsigned short As[128 * 40];
  __shared__ __attribute__((aligned(16))) unsigned short Bs[32 * 72];
  const int ASTR = 40, BSTR = 72;  // rows 80B / 144B => 16B-aligned chunks
  const int row_base = blockIdx.x * 128;
  const int col_base = blockIdx.y * 64;
  const int wave = threadIdx.x >> 5;
  const int wm = (wave >> 1) * 32;
  const int wn = (wave & 1) * 32;
  const int tid = threadIdx.x;
  v8f acc[2][2];
  acc[0][0] = zero8(); acc[0][1] = zero8(); acc[1][0] = zero8(); acc[1][1] = zero8();

  for (int k0 = 0; k0 < K; k0 += 32) {
    // ---- stage A tile: 128x32 bf16 ----
    if (A32) {
      // fp32 -> bf16 convert in flight (16 elems/thread)
      for (int i = tid; i < 128 * 32; i += 256) {
        int r = i >> 5, cc = i & 31;
        As[r * ASTR + cc] = f2bf(A32[(size_t)(row_base + r) * K + k0 + cc]);
      }
      if (k0 + 32 < K) {  // prefetch next A tile (global_prefetch_b8)
        int r = tid >> 1, cc = (tid & 1) * 16;
        __builtin_prefetch(&A32[(size_t)(row_base + r) * K + k0 + 32 + cc], 0, 1);
      }
    } else {
#if HAS_ASYNC_LDS
      // 128 rows * 64B = 512 x 16B chunks, 2 per thread, VGPR-free copy
#pragma unroll
      for (int j = 0; j < 2; ++j) {
        int idx = tid + j * 256;
        int r = idx >> 2, cc = (idx & 3) * 8;
        async_g2l_b128(&A16[(size_t)(row_base + r) * K + k0 + cc], &As[r * ASTR + cc]);
      }
#else
      for (int i = tid; i < 128 * 32; i += 256) {
        int r = i >> 5, cc = i & 31;
        As[r * ASTR + cc] = A16[(size_t)(row_base + r) * K + k0 + cc];
      }
#endif
    }
    // ---- stage B tile: 32x64 bf16 = 256 x 16B chunks, 1 per thread ----
#if HAS_ASYNC_LDS
    {
      int r = tid >> 3, cc = (tid & 7) * 8;
      async_g2l_b128(&B[(size_t)(k0 + r) * Nout + col_base + cc], &Bs[r * BSTR + cc]);
    }
    wait_async0();
#else
    for (int i = tid; i < 32 * 64; i += 256) {
      int r = i >> 6, cc = i & 63;
      Bs[r * BSTR + cc] = B[(size_t)(k0 + r) * Nout + col_base + cc];
    }
#endif
    __syncthreads();
    v16bf a0 = frag_rowmajor(As, wm, ASTR);
    v16bf a1 = frag_rowmajor(As, wm + 16, ASTR);
    v16bf b0 = frag_colmajor(Bs, wn, BSTR);
    v16bf b1 = frag_colmajor(Bs, wn + 16, BSTR);
    acc[0][0] = wmma_bf16(a0, b0, acc[0][0]);
    acc[0][1] = wmma_bf16(a0, b1, acc[0][1]);
    acc[1][0] = wmma_bf16(a1, b0, acc[1][0]);
    acc[1][1] = wmma_bf16(a1, b1, acc[1][1]);
    __syncthreads();
  }
  const int lane = threadIdx.x & 31;
  const int hi = (lane >> 4) & 1;
#pragma unroll
  for (int mi = 0; mi < 2; ++mi)
#pragma unroll
    for (int ni = 0; ni < 2; ++ni) {
      int col = col_base + wn + ni * 16 + (lane & 15);
      float bv = bias ? bias[col] : 0.f;
#pragma unroll
      for (int r = 0; r < 8; ++r) {
        int row = row_base + wm + mi * 16 + hi * 8 + r;
        C[(size_t)row * Nout + col] = acc[mi][ni][r] + bv;
      }
    }
}

// ---------- per-channel stats (sum, sumsq) ----------
__global__ void stats_k(const float* __restrict__ X, size_t total, int cols, int nch,
                        float* __restrict__ stats) {
  __shared__ float ssum[32];
  __shared__ float ssq[32];
  if (threadIdx.x < 32) { ssum[threadIdx.x] = 0.f; ssq[threadIdx.x] = 0.f; }
  __syncthreads();
  for (size_t i = (size_t)blockIdx.x * blockDim.x + threadIdx.x; i < total;
       i += (size_t)gridDim.x * blockDim.x) {
    float v = X[i];
    int ch = (int)((i / (size_t)cols) % (size_t)nch);
    atomicAdd(&ssum[ch], v);
    atomicAdd(&ssq[ch], v * v);
  }
  __syncthreads();
  if ((int)threadIdx.x < nch) {
    atomicAdd(&stats[threadIdx.x], ssum[threadIdx.x]);
    atomicAdd(&stats[nch + threadIdx.x], ssq[threadIdx.x]);
  }
}

// ---------- per-sample attention: bn0 -> G=V.V^T -> softsign -> V'=G.V ----------
__global__ void attn_k(const float* __restrict__ Y0, const float* __restrict__ stats,
                       const float* __restrict__ g, const float* __restrict__ b,
                       float denom, unsigned short* __restrict__ Vp) {
  __shared__ __attribute__((aligned(16))) unsigned short Vs[32 * 264];
  __shared__ __attribute__((aligned(16))) unsigned short Ws[32 * 40];
  const int s = blockIdx.x;
  for (int i = threadIdx.x; i < 32 * 256; i += 256) {
    int c = i >> 8, d = i & 255;
    float v = 0.f;
    if (c < NFEAT) {
      float m = stats[c] / denom;
      float var = stats[NFEAT + c] / denom - m * m;
      v = (Y0[((size_t)s * NFEAT + c) * HIDV + d] - m) * rsqrtf(var + EPSV) * g[c] + b[c];
    }
    Vs[c * 264 + d] = f2bf(v);
  }
  __syncthreads();

  const int wave = threadIdx.x >> 5;
  const int lane = threadIdx.x & 31;
  const int hi = (lane >> 4) & 1;
  if (wave < 4) {  // G tiles: 2x2 of 16x16 over K=256
    const int wm = (wave >> 1) * 16;
    const int wn = (wave & 1) * 16;
    v8f acc = zero8();
    for (int k0 = 0; k0 < 256; k0 += 32) {
      v16bf a = frag_rowmajor(&Vs[k0], wm, 264);
      v16bf bt = frag_rowmajor(&Vs[k0], wn, 264);  // B[k][n]=V[n][k] => row addressing
      acc = wmma_bf16(a, bt, acc);
    }
#pragma unroll
    for (int r = 0; r < 8; ++r) {
      int mrow = wm + hi * 8 + r;
      int ncol = wn + (lane & 15);
      Ws[mrow * 40 + ncol] = f2bf(ssign(acc[r]));
    }
  }
  __syncthreads();

  // V' = W(32x32) * V(32x256): 2x16 tiles, 8 waves * 4 tiles each
#pragma unroll
  for (int j = 0; j < 4; ++j) {
    int t = wave * 4 + j;
    int tm = t & 1;
    int tn = t >> 1;
    v16bf a = frag_rowmajor(Ws, tm * 16, 40);
    v16bf bb = frag_colmajor(Vs, tn * 16, 264);
    v8f acc = zero8();
    acc = wmma_bf16(a, bb, acc);
#pragma unroll
    for (int r = 0; r < 8; ++r) {
      int row = tm * 16 + hi * 8 + r;
      int col = tn * 16 + (lane & 15);
      if (row < NFEAT)
        Vp[((size_t)s * NFEAT + row) * HIDV + col] = f2bf(acc[r]);
    }
  }
}

// ---------- elementwise: T = bn1(T) + X ----------
__global__ void bnadd_k(float* __restrict__ T, const float* __restrict__ X,
                        const float* __restrict__ stats, const float* __restrict__ g,
                        const float* __restrict__ b, float denom, size_t total) {
  for (size_t i = (size_t)blockIdx.x * blockDim.x + threadIdx.x; i < total;
       i += (size_t)gridDim.x * blockDim.x) {
    int ch = (int)((i / DIMV) % NFEAT);
    float m = stats[ch] / denom;
    float var = stats[NFEAT + ch] / denom - m * m;
    T[i] = (T[i] - m) * rsqrtf(var + EPSV) * g[ch] + b[ch] + X[i];
  }
}

// ---------- per-sample token mix ----------
__global__ void token_mix_k(const float* __restrict__ In, const float* __restrict__ stats,
                            const float* __restrict__ g, const float* __restrict__ b,
                            float denom, const float* __restrict__ W,
                            const float* __restrict__ bias, int addres,
                            float* __restrict__ Out) {
  __shared__ float Xs[NFEAT * 516];
  const int s = blockIdx.x;
  for (int i = threadIdx.x; i < NFEAT * DIMV; i += 256) {
    int c = i / DIMV, d = i % DIMV;
    float v = In[((size_t)s * NFEAT + c) * DIMV + d];
    if (stats) {
      float m = stats[c] / denom;
      float var = stats[NFEAT + c] / denom - m * m;
      v = (v - m) * rsqrtf(var + EPSV) * g[c] + b[c];
    }
    Xs[c * 516 + d] = v;
  }
  __syncthreads();
  for (int o = threadIdx.x; o < NFEAT * DIMV; o += 256) {
    int c = o / DIMV, d = o % DIMV;
    float acc = bias[c];
#pragma unroll
    for (int k = 0; k < NFEAT; ++k) acc += W[c * NFEAT + k] * Xs[k * 516 + d];
    if (addres) acc += Xs[c * 516 + d];
    Out[((size_t)s * NFEAT + c) * DIMV + d] = acc;
  }
}

// ---------- elementwise bn apply (+softsign, +Z accumulate) ----------
__global__ void bn_apply_k(const float* __restrict__ In, float* __restrict__ Out,
                           const float* __restrict__ stats, const float* __restrict__ g,
                           const float* __restrict__ b, float denom, int nch, int cols,
                           int do_ss, int zmode, float* __restrict__ Z, size_t total) {
  for (size_t i = (size_t)blockIdx.x * blockDim.x + threadIdx.x; i < total;
       i += (size_t)gridDim.x * blockDim.x) {
    int ch = (int)((i / (size_t)cols) % (size_t)nch);
    float m = stats[ch] / denom;
    float var = stats[nch + ch] / denom - m * m;
    float v = (In[i] - m) * rsqrtf(var + EPSV) * g[ch] + b[ch];
    if (do_ss) v = ssign(v);
    Out[i] = v;
    if (zmode == 1) Z[i] = v;
    else if (zmode == 2) Z[i] += v;
  }
}

// ---------- head: vv = sum_k c2W[k] * Z[n,k,:] + c2b ----------
__global__ void c2_k(const float* __restrict__ Z, const float* __restrict__ w,
                     const float* __restrict__ cb, float* __restrict__ VV, size_t total) {
  for (size_t i = (size_t)blockIdx.x * blockDim.x + threadIdx.x; i < total;
       i += (size_t)gridDim.x * blockDim.x) {
    size_t n = i / DIMV;
    int d = (int)(i % DIMV);
    float acc = cb[0];
#pragma unroll
    for (int k = 0; k < NFEAT; ++k) acc += w[k] * Z[(n * NFEAT + k) * DIMV + d];
    VV[i] = acc;
  }
}

// ---------- head: out[n] = dot(vv[n,:], fcW) + fcb ----------
__global__ void fc_k(const float* __restrict__ VV, const float* __restrict__ w,
                     const float* __restrict__ bp, float* __restrict__ O) {
  __shared__ float red[256];
  const int n = blockIdx.x;
  float s = 0.f;
  for (int d = threadIdx.x; d < DIMV; d += 256)
    s += VV[(size_t)n * DIMV + d] * w[d];
  red[threadIdx.x] = s;
  __syncthreads();
  for (int st = 128; st > 0; st >>= 1) {
    if ((int)threadIdx.x < st) red[threadIdx.x] += red[threadIdx.x + st];
    __syncthreads();
  }
  if (threadIdx.x == 0) O[n] = red[0] + bp[0];
}

// =====================================================================
extern "C" void kernel_launch(void* const* d_in, const int* in_sizes, int n_in,
                              void* d_out, int out_size, void* d_ws, size_t ws_size,
                              hipStream_t stream) {
  (void)in_sizes; (void)n_in; (void)out_size; (void)ws_size;
  const int* xin       = (const int*)d_in[0];
  const float* emb     = (const float*)d_in[1];
  const float* loc     = (const float*)d_in[2];
  const float* ln_g    = (const float*)d_in[3];
  const float* ln_b    = (const float*)d_in[4];
  const float* W0      = (const float*)d_in[5];
  const float* b0      = (const float*)d_in[6];
  const float* bn0_g   = (const float*)d_in[7];
  const float* bn0_b   = (const float*)d_in[8];
  const float* W1      = (const float*)d_in[9];
  const float* b1      = (const float*)d_in[10];
  const float* bn1_g   = (const float*)d_in[11];
  const float* bn1_b   = (const float*)d_in[12];
  const float* feed_g  = (const float*)d_in[13];
  const float* feed_b  = (const float*)d_in[14];
  const float* convW   = (const float*)d_in[15];
  const float* convb   = (const float*)d_in[16];
  const float* out_g   = (const float*)d_in[17];
  const float* out_b   = (const float*)d_in[18];
  const float* c0_W    = (const float*)d_in[19];
  const float* c0_b    = (const float*)d_in[20];
  const float* c0_g    = (const float*)d_in[21];
  const float* c0_bb   = (const float*)d_in[22];
  const float* c1_W    = (const float*)d_in[23];
  const float* c1_b    = (const float*)d_in[24];
  const float* c1_g    = (const float*)d_in[25];
  const float* c1_bb   = (const float*)d_in[26];
  const float* c2_W    = (const float*)d_in[27];
  const float* c2_b    = (const float*)d_in[28];
  const float* c2_g    = (const float*)d_in[29];
  const float* c2_bb   = (const float*)d_in[30];
  const float* fc_W    = (const float*)d_in[31];
  const float* fc_b    = (const float*)d_in[32];
  const float* fc_g    = (const float*)d_in[33];
  const float* fc_bb   = (const float*)d_in[34];

  const int M = BATCHV * NFEAT;                 // 102400
  const size_t T512 = (size_t)M * DIMV;
  const size_t T256 = (size_t)M * HIDV;

  char* ws = (char*)d_ws;
  size_t off = 0;
  float* X  = (float*)(ws + off); off += T512 * 4;
  float* Y0 = (float*)(ws + off); off += T256 * 4;
  unsigned short* Vp = (unsigned short*)(ws + off); off += T256 * 2;
  float* T  = (float*)(ws + off); off += T512 * 4;
  float* R  = (float*)(ws + off); off += T512 * 4;
  float* Z  = (float*)(ws + off); off += T512 * 4;
  float* VV = (float*)(ws + off); off += (size_t)BATCHV * DIMV * 4;
  float* OO = (float*)(ws + off); off += (size_t)BATCHV * 4;
  unsigned short* W0b = (unsigned short*)(ws + off); off += (size_t)NBLK * DIMV * HIDV * 2;
  unsigned short* W1b = (unsigned short*)(ws + off); off += (size_t)NBLK * HIDV * DIMV * 2;
  float* ST = (float*)(ws + off); off += 64 * 4;

  const float DEN_H = (float)BATCHV * HIDV;
  const float DEN_D = (float)BATCHV * DIMV;

  const dim3 blk256(256);
  const int EW_GRID = 8192;
  const int ST_GRID = 2048;

  pack_k<<<dim3(2048), blk256, 0, stream>>>(W0, W1, W0b, W1b, (size_t)NBLK * DIMV * HIDV);
  embed_k<<<dim3(M), blk256, 0, stream>>>(xin, emb, loc, ln_g, ln_b, X);

  for (int blkid = 0; blkid < NBLK; ++blkid) {
    const unsigned short* w0p = W0b + (size_t)blkid * DIMV * HIDV;
    const unsigned short* w1p = W1b + (size_t)blkid * HIDV * DIMV;
    const float* b0p = b0 + blkid * HIDV;
    const float* b1p = b1 + blkid * DIMV;

    gemm_k<<<dim3(M / 128, HIDV / 64), blk256, 0, stream>>>(X, nullptr, w0p, b0p, Y0, M, DIMV, HIDV);
    zero_k<<<1, 64, 0, stream>>>(ST, 64);
    stats_k<<<dim3(ST_GRID), blk256, 0, stream>>>(Y0, T256, HIDV, NFEAT, ST);
    attn_k<<<dim3(BATCHV), blk256, 0, stream>>>(Y0, ST, bn0_g + blkid * NFEAT,
                                                bn0_b + blkid * NFEAT, DEN_H, Vp);
    gemm_k<<<dim3(M / 128, DIMV / 64), blk256, 0, stream>>>(nullptr, Vp, w1p, b1p, T, M, HIDV, DIMV);
    zero_k<<<1, 64, 0, stream>>>(ST, 64);
    stats_k<<<dim3(ST_GRID), blk256, 0, stream>>>(T, T512, DIMV, NFEAT, ST);
    bnadd_k<<<dim3(EW_GRID), blk256, 0, stream>>>(T, X, ST, bn1_g + blkid * NFEAT,
                                                  bn1_b + blkid * NFEAT, DEN_D, T512);
    zero_k<<<1, 64, 0, stream>>>(ST, 64);
    stats_k<<<dim3(ST_GRID), blk256, 0, stream>>>(T, T512, DIMV, NFEAT, ST);
    token_mix_k<<<dim3(BATCHV), blk256, 0, stream>>>(T, ST, feed_g + blkid * NFEAT,
                                                     feed_b + blkid * NFEAT, DEN_D,
                                                     convW + (size_t)blkid * NFEAT * NFEAT,
                                                     convb + blkid * NFEAT, 1, R);
    zero_k<<<1, 64, 0, stream>>>(ST, 64);
    stats_k<<<dim3(ST_GRID), blk256, 0, stream>>>(R, T512, DIMV, NFEAT, ST);
    int zmode = (blkid == 2) ? 1 : ((blkid == 7 || blkid == 10) ? 2 : 0);
    bn_apply_k<<<dim3(EW_GRID), blk256, 0, stream>>>(R, X, ST, out_g + blkid * NFEAT,
                                                     out_b + blkid * NFEAT, DEN_D, NFEAT,
                                                     DIMV, 0, zmode, Z, T512);
    if (blkid == 2 || blkid == 7) {
      const float* hW = (blkid == 2) ? c0_W : c1_W;
      const float* hb = (blkid == 2) ? c0_b : c1_b;
      const float* hg = (blkid == 2) ? c0_g : c1_g;
      const float* hbb = (blkid == 2) ? c0_bb : c1_bb;
      token_mix_k<<<dim3(BATCHV), blk256, 0, stream>>>(X, nullptr, nullptr, nullptr, 1.f,
                                                       hW, hb, 0, T);
      zero_k<<<1, 64, 0, stream>>>(ST, 64);
      stats_k<<<dim3(ST_GRID), blk256, 0, stream>>>(T, T512, DIMV, NFEAT, ST);
      bn_apply_k<<<dim3(EW_GRID), blk256, 0, stream>>>(T, X, ST, hg, hbb, DEN_D, NFEAT,
                                                       DIMV, 1, 0, nullptr, T512);
    }
  }

  const size_t TVV = (size_t)BATCHV * DIMV;
  c2_k<<<dim3(4096), blk256, 0, stream>>>(Z, c2_W, c2_b, VV, TVV);
  zero_k<<<1, 64, 0, stream>>>(ST, 64);
  stats_k<<<dim3(ST_GRID), blk256, 0, stream>>>(VV, TVV, DIMV, 1, ST);
  bn_apply_k<<<dim3(4096), blk256, 0, stream>>>(VV, VV, ST, c2_g, c2_bb,
                                                (float)BATCHV * DIMV, 1, DIMV, 1, 0,
                                                nullptr, TVV);
  fc_k<<<dim3(BATCHV), blk256, 0, stream>>>(VV, fc_W, fc_b, OO);
  zero_k<<<1, 64, 0, stream>>>(ST, 64);
  stats_k<<<dim3(64), blk256, 0, stream>>>(OO, (size_t)BATCHV, 1, 1, ST);
  bn_apply_k<<<dim3(16), blk256, 0, stream>>>(OO, (float*)d_out, ST, fc_g, fc_bb,
                                              (float)BATCHV, 1, 1, 0, 0, nullptr,
                                              (size_t)BATCHV);
}
